// AKWA_10118942949711
// MI455X (gfx1250) — compile-verified, matrix operand order checked by reading, not run on hardware
//
#include <hip/hip_runtime.h>
#include <hip/hip_bf16.h>
#include <math.h>

#define B_  8
#define C_  256
#define C4_ 64
#define HD_ 4
#define N_  16384   // 128*128

typedef __bf16 bf16;
typedef __attribute__((ext_vector_type(16))) __bf16 v16bf;
typedef __attribute__((ext_vector_type(8)))  float  v8f;

union FragU { uint4 q[2]; v16bf v; };

// CDNA5 16-bit A(16x32)/B(32x16) fragments share one lane layout:
//   i = lane&15 (row for A / col for B), element j -> k = (j&7) + (j>=8?16:0) + (lane>=16?8:0)
// => per lane: two contiguous 8-element (16B) runs at k0 and k0+16, k0 = (lane&16)>>1.
__device__ __forceinline__ v16bf frag_load(const bf16* p, int lane) {
  FragU f;
  const bf16* pp = p + ((lane & 16) >> 1);
  f.q[0] = *(const uint4*)(pp);
  f.q[1] = *(const uint4*)(pp + 16);
  return f.v;
}
__device__ __forceinline__ v16bf frag_zero() {
  FragU f;
  f.q[0] = make_uint4(0u,0u,0u,0u);
  f.q[1] = make_uint4(0u,0u,0u,0u);
  return f.v;
}
__device__ __forceinline__ v8f wmma_bf16(v16bf a, v16bf b, v8f c) {
  return __builtin_amdgcn_wmma_f32_16x16x32_bf16(false, a, false, b, (short)0, c,
                                                 false, false);
}

// ---- 1) query (b,c,N) f32 -> X (b,N,c) bf16, LDS-tiled transpose ----------
__global__ void k_x_bf16_T(const float* __restrict__ q, bf16* __restrict__ X) {
  __shared__ bf16 tile[32][33];
  int blk = blockIdx.x;
  int b   = blk >> 12;          // 4096 tiles per batch (512 n-tiles * 8 c-tiles)
  int rem = blk & 4095;
  int n0  = (rem >> 3) << 5;
  int c0  = (rem & 7) << 5;
  int tx  = threadIdx.x & 31, ty = threadIdx.x >> 5;   // ty: 0..7
  #pragma unroll
  for (int i = 0; i < 4; ++i) {
    int cc = ty + (i << 3);
    tile[cc][tx] = (bf16)q[((size_t)(b * C_ + c0 + cc)) * N_ + n0 + tx];
  }
  __syncthreads();
  #pragma unroll
  for (int i = 0; i < 4; ++i) {
    int nn = ty + (i << 3);
    X[((size_t)(b * N_ + n0 + nn)) * C_ + c0 + tx] = tile[tx][nn];
  }
}

// ---- 2) weights -> bf16; Wk reordered to (h, tap, o, c) -------------------
__global__ void k_weights(const float* __restrict__ Wq, const float* __restrict__ Wv,
                          const float* __restrict__ Wk, bf16* __restrict__ Wqb,
                          bf16* __restrict__ Wvb, bf16* __restrict__ Wkb) {
  int t = blockIdx.x * 256 + threadIdx.x;
  if (t < 65536) { Wqb[t] = (bf16)Wq[t]; return; }
  t -= 65536;
  if (t < 65536) { Wvb[t] = (bf16)Wv[t]; return; }
  t -= 65536;
  if (t < 589824) {                 // idx = ((h*9+tap)<<14) + (o<<8) + cc
    int cc = t & 255, o = (t >> 8) & 63, top = t >> 14;
    int h = top / 9, tap = top % 9;
    Wkb[t] = (bf16)Wk[(((size_t)(h * 64 + o) * 256 + cc) * 9) + tap];
  }
}

// ---- 3) fused Q (c4,N) and V (N,c4) projections; X frag reused as B and A -
__global__ void k_qvproj(const bf16* __restrict__ X, const bf16* __restrict__ Wqb,
                         const bf16* __restrict__ Wvb, const float* __restrict__ bq,
                         const float* __restrict__ bv, bf16* __restrict__ Q,
                         bf16* __restrict__ V) {
  int lane  = threadIdx.x & 31;
  int gw    = blockIdx.x * 8 + (threadIdx.x >> 5);     // 32768 waves
  int ntile = gw & 1023;
  int hb    = gw >> 10;                                // h*8+b
  int h = hb >> 3, b = hb & 7;
  int col   = lane & 15;
  int nbase = ntile << 4;

  const bf16* xp  = X   + ((size_t)(b * N_ + nbase + col)) * C_;
  const bf16* wq0 = Wqb + ((size_t)(h * 64 + col)) * C_;
  const bf16* wv0 = Wvb + ((size_t)(h * 64 + col)) * C_;

  v8f accQ[4] = { {}, {}, {}, {} };
  v8f accV[4] = { {}, {}, {}, {} };

  for (int cs = 0; cs < 8; ++cs) {
    int kb = cs << 5;
    v16bf xf = frag_load(xp + kb, lane);               // shared X fragment
    #pragma unroll
    for (int ot = 0; ot < 4; ++ot) {
      v16bf a = frag_load(wq0 + (size_t)(ot * 16) * C_ + kb, lane);
      accQ[ot] = wmma_bf16(a, xf, accQ[ot]);           // Q = Wq * X
      v16bf w = frag_load(wv0 + (size_t)(ot * 16) * C_ + kb, lane);
      accV[ot] = wmma_bf16(xf, w, accV[ot]);           // V^T = X^T * Wv^T
    }
  }
  int mAdd = (lane >> 4) << 3;
  #pragma unroll
  for (int ot = 0; ot < 4; ++ot)
    #pragma unroll
    for (int r = 0; r < 8; ++r) {
      int o = ot * 16 + r + mAdd;
      Q[((size_t)(hb * 64 + o)) * N_ + nbase + col] = (bf16)(accQ[ot][r] + bq[h * 64 + o]);
    }
  #pragma unroll
  for (int ot = 0; ot < 4; ++ot) {
    int o = ot * 16 + col;
    float bias = bv[h * 64 + o];
    #pragma unroll
    for (int r = 0; r < 8; ++r) {
      int tok = nbase + r + mAdd;
      V[((size_t)(hb * N_ + tok)) * 64 + o] = (bf16)(accV[ot][r] + bias);
    }
  }
}

// ---- 4) dilated 3x3 key conv as implicit GEMM (dilation = 1<<head) --------
// One wave computes a 64(o) x 32(n) output tile: each Wk fragment is loaded
// once and feeds two WMMAs (two n-tiles) -> 1.5 B128 loads per WMMA.
__global__ void k_keyconv(const bf16* __restrict__ X, const bf16* __restrict__ Wkb,
                          const float* __restrict__ bk, bf16* __restrict__ K) {
  int lane  = threadIdx.x & 31;
  int gw    = blockIdx.x * 8 + (threadIdx.x >> 5);     // 16384 waves
  int npair = gw & 511;                                // 512 pairs of n-tiles
  int hb    = gw >> 9;
  int h = hb >> 3, b = hb & 7;
  int col   = lane & 15;
  int nbase = npair << 5;                              // 32 tokens
  int n0 = nbase + col, n1 = n0 + 16;
  int y0 = n0 >> 7, x0 = n0 & 127;
  int y1 = n1 >> 7, x1 = n1 & 127;
  int d = 1 << h;

  v8f acc[4][2] = { { {}, {} }, { {}, {} }, { {}, {} }, { {}, {} } };
  const bf16* wk0 = Wkb + ((size_t)h * 9 * 64) * C_ + (size_t)col * C_;

  for (int tap = 0; tap < 9; ++tap) {
    int dy = ((tap / 3) - 1) * d, dx = ((tap % 3) - 1) * d;
    int yy0 = y0 + dy, xx0 = x0 + dx;
    int yy1 = y1 + dy, xx1 = x1 + dx;
    bool ok0 = ((unsigned)yy0 < 128u) && ((unsigned)xx0 < 128u);
    bool ok1 = ((unsigned)yy1 < 128u) && ((unsigned)xx1 < 128u);
    int t0 = ok0 ? ((yy0 << 7) + xx0) : 0;
    int t1 = ok1 ? ((yy1 << 7) + xx1) : 0;
    const bf16* xp0 = X + ((size_t)(b * N_ + t0)) * C_;
    const bf16* xp1 = X + ((size_t)(b * N_ + t1)) * C_;
    const bf16* wt  = wk0 + (size_t)tap * 64 * C_;
    __builtin_prefetch((const void*)xp0, 0, 0);
    __builtin_prefetch((const void*)xp1, 0, 0);
    for (int cs = 0; cs < 8; ++cs) {
      int kb = cs << 5;
      v16bf xf0 = ok0 ? frag_load(xp0 + kb, lane) : frag_zero();
      v16bf xf1 = ok1 ? frag_load(xp1 + kb, lane) : frag_zero();
      #pragma unroll
      for (int ot = 0; ot < 4; ++ot) {
        v16bf a = frag_load(wt + (size_t)(ot * 16) * C_ + kb, lane);
        acc[ot][0] = wmma_bf16(a, xf0, acc[ot][0]);    // weight frag reused
        acc[ot][1] = wmma_bf16(a, xf1, acc[ot][1]);
      }
    }
  }
  int mAdd = (lane >> 4) << 3;
  #pragma unroll
  for (int ot = 0; ot < 4; ++ot)
    #pragma unroll
    for (int r = 0; r < 8; ++r) {
      int o = ot * 16 + r + mAdd;
      float bias = bk[h * 64 + o];
      size_t base = ((size_t)(hb * 64 + o)) * N_ + nbase + col;
      K[base]      = (bf16)(acc[ot][0][r] + bias);
      K[base + 16] = (bf16)(acc[ot][1][r] + bias);
    }
}

// ---- 5) scores = scale * Q K^T (contract N); 8-wave split-K + LDS reduce --
__global__ void k_scores(const bf16* __restrict__ Q, const bf16* __restrict__ K,
                         float* __restrict__ S) {
  __shared__ float red[8][256];
  int lane = threadIdx.x & 31;
  int wv   = threadIdx.x >> 5;
  int blk  = blockIdx.x;                 // 512 = 32 hb * 4 ot * 4 pt
  int pt = blk & 3, ot = (blk >> 2) & 3, hb = blk >> 4;
  int col = lane & 15;

  const bf16* qp = Q + ((size_t)(hb * 64 + ot * 16 + col)) * N_;
  const bf16* kp = K + ((size_t)(hb * 64 + pt * 16 + col)) * N_;

  v8f acc = {};
  int n0 = wv << 11;                     // 2048 tokens per wave
  for (int s = 0; s < 64; ++s) {
    int nb = n0 + (s << 5);
    v16bf a  = frag_load(qp + nb, lane);
    v16bf bb = frag_load(kp + nb, lane);
    acc = wmma_bf16(a, bb, acc);
  }
  int mAdd = (lane >> 4) << 3;
  #pragma unroll
  for (int r = 0; r < 8; ++r)
    red[wv][(r + mAdd) * 16 + col] = acc[r];
  __syncthreads();
  int t = threadIdx.x;
  float sum = 0.f;
  #pragma unroll
  for (int w = 0; w < 8; ++w) sum += red[w][t];
  int m = t >> 4, cc = t & 15;
  S[((size_t)hb * 64 + ot * 16 + m) * 64 + pt * 16 + cc] = sum * 0.0625f; // 1/sqrt(256)
}

// ---- 6) softmax over 64 (one wave per row), attn -> bf16 ------------------
__global__ void k_softmax(const float* __restrict__ S, bf16* __restrict__ A) {
  int row = blockIdx.x, lane = threadIdx.x;
  float s0 = S[row * 64 + lane], s1 = S[row * 64 + 32 + lane];
  float m = fmaxf(s0, s1);
  for (int off = 16; off > 0; off >>= 1) m = fmaxf(m, __shfl_xor(m, off, 32));
  float e0 = __expf(s0 - m), e1 = __expf(s1 - m);
  float sum = e0 + e1;
  for (int off = 16; off > 0; off >>= 1) sum += __shfl_xor(sum, off, 32);
  float inv = 1.0f / sum;
  A[row * 64 + lane]      = (bf16)(e0 * inv);
  A[row * 64 + 32 + lane] = (bf16)(e1 * inv);
}

// ---- 7) out = attn @ V, written f32 into (b, H*c4, h, w) ------------------
__global__ void k_out(const bf16* __restrict__ A, const bf16* __restrict__ V,
                      float* __restrict__ out) {
  int lane  = threadIdx.x & 31;
  int gw    = blockIdx.x * 8 + (threadIdx.x >> 5);     // 16384 waves
  int chunk = gw & 127;
  int ot    = (gw >> 7) & 3;
  int hb    = gw >> 9;
  int h = hb >> 3, b = hb & 7;
  int col = lane & 15;

  const bf16* ap = A + (size_t)hb * 4096 + (size_t)(ot * 16 + col) * 64;
  v16bf a0 = frag_load(ap, lane);        // K(p)=0..31
  v16bf a1 = frag_load(ap + 32, lane);   // K(p)=32..63

  int mAdd = (lane >> 4) << 3;
  for (int t = 0; t < 8; ++t) {
    int nbase = (chunk * 8 + t) << 4;
    const bf16* bp = V + ((size_t)(hb * N_ + nbase + col)) * 64;
    v16bf b0 = frag_load(bp, lane);
    v16bf b1 = frag_load(bp + 32, lane);
    v8f acc = {};
    acc = wmma_bf16(a0, b0, acc);
    acc = wmma_bf16(a1, b1, acc);
    #pragma unroll
    for (int r = 0; r < 8; ++r) {
      int o = ot * 16 + r + mAdd;
      out[((size_t)(b * 256 + h * 64 + o)) * N_ + nbase + col] = acc[r];
    }
  }
}

extern "C" void kernel_launch(void* const* d_in, const int* in_sizes, int n_in,
                              void* d_out, int out_size, void* d_ws, size_t ws_size,
                              hipStream_t stream) {
  (void)in_sizes; (void)n_in; (void)out_size; (void)ws_size;
  const float* query = (const float*)d_in[0];
  const float* Wq    = (const float*)d_in[1];
  const float* bq    = (const float*)d_in[2];
  const float* Wk    = (const float*)d_in[3];
  const float* bk    = (const float*)d_in[4];
  const float* Wv    = (const float*)d_in[5];
  const float* bv    = (const float*)d_in[6];
  float* out = (float*)d_out;

  char* ws = (char*)d_ws;
  size_t off = 0;
  bf16*  Xbf = (bf16*)(ws + off); off += (size_t)B_ * N_ * C_ * 2;    // 64 MiB
  bf16*  Wqb = (bf16*)(ws + off); off += (size_t)65536 * 2;
  bf16*  Wvb = (bf16*)(ws + off); off += (size_t)65536 * 2;
  bf16*  Wkb = (bf16*)(ws + off); off += (size_t)589824 * 2;
  bf16*  Qb  = (bf16*)(ws + off); off += (size_t)32 * 64 * N_ * 2;    // 64 MiB
  bf16*  Kb  = (bf16*)(ws + off); off += (size_t)32 * 64 * N_ * 2;    // 64 MiB
  bf16*  Vb  = (bf16*)(ws + off); off += (size_t)32 * N_ * 64 * 2;    // 64 MiB
  float* Sc  = (float*)(ws + off); off += (size_t)32 * 64 * 64 * 4;
  bf16*  At  = (bf16*)(ws + off); off += (size_t)32 * 64 * 64 * 2;

  hipLaunchKernelGGL(k_x_bf16_T, dim3(32768), dim3(256), 0, stream, query, Xbf);
  hipLaunchKernelGGL(k_weights,  dim3(2816),  dim3(256), 0, stream, Wq, Wv, Wk, Wqb, Wvb, Wkb);
  hipLaunchKernelGGL(k_qvproj,   dim3(4096),  dim3(256), 0, stream, Xbf, Wqb, Wvb, bq, bv, Qb, Vb);
  hipLaunchKernelGGL(k_keyconv,  dim3(2048),  dim3(256), 0, stream, Xbf, Wkb, bk, Kb);
  hipLaunchKernelGGL(k_scores,   dim3(512),   dim3(256), 0, stream, Qb, Kb, Sc);
  hipLaunchKernelGGL(k_softmax,  dim3(2048),  dim3(32),  0, stream, Sc, At);
  hipLaunchKernelGGL(k_out,      dim3(2048),  dim3(256), 0, stream, At, Vb, out);
}
